// Attention_455266533383
// MI455X (gfx1250) — matrix operand order repaired and checked by reading, not compiled, not run on hardware
//
#include <hip/hip_runtime.h>
#include <hip/hip_bf16.h>
#include <math.h>

// ---------------------------------------------------------------------------
// Types for CDNA5 WMMA (gfx1250, wave32)
// ---------------------------------------------------------------------------
typedef __attribute__((ext_vector_type(16))) _Float16 v16h;
typedef __attribute__((ext_vector_type(8)))  _Float16 v8h;
typedef __attribute__((ext_vector_type(8)))  float    v8f;
typedef int v4i __attribute__((vector_size(4 * sizeof(int))));

union V16 { v16h v; v8h h[2]; };

__device__ __forceinline__ v8f wmma_f16(v16h a, v16h b, v8f c) {
    // D = A(16x32 f16) * B(32x16 f16) + C(16x16 f32)
    return __builtin_amdgcn_wmma_f32_16x16x32_f16(
        /*neg_a=*/false, a, /*neg_b=*/false, b,
        /*c_mod=*/(short)0, c, /*reuse_a=*/false, /*reuse_b=*/false);
}

// ---------------------------------------------------------------------------
// Async global->LDS copy path (CDNA5). Falls back to sync copy if the
// builtins aren't available on this toolchain.
// ---------------------------------------------------------------------------
#if __has_builtin(__builtin_amdgcn_global_load_async_to_lds_b128) && \
    __has_builtin(__builtin_amdgcn_s_wait_asynccnt)
#define USE_ASYNC_LDS 1
#else
#define USE_ASYNC_LDS 0
#endif

__device__ __forceinline__ void cp16_g2s(const _Float16* g, _Float16* l) {
#if USE_ASYNC_LDS
    __builtin_amdgcn_global_load_async_to_lds_b128(
        (__attribute__((address_space(1))) v4i*)g,
        (__attribute__((address_space(3))) v4i*)l, 0, 0);
#else
    *(v8h*)l = *(const v8h*)g;
#endif
}

__device__ __forceinline__ void async_wait_4() {
#if USE_ASYNC_LDS
    __builtin_amdgcn_s_wait_asynccnt(4);
#endif
}
__device__ __forceinline__ void async_wait_0() {
#if USE_ASYNC_LDS
    __builtin_amdgcn_s_wait_asynccnt(0);
#endif
}

// ---------------------------------------------------------------------------
// Elementwise f32 -> f16 conversion
// ---------------------------------------------------------------------------
__global__ void cvt_f32_f16(const float* __restrict__ src,
                            _Float16* __restrict__ dst, int n) {
    int i = blockIdx.x * blockDim.x + threadIdx.x;
    int stride = gridDim.x * blockDim.x;
    for (; i < n; i += stride) dst[i] = (_Float16)src[i];
}

// ---------------------------------------------------------------------------
// Block-tiled GEMM: C[M,N] (f32) = A[M,K] (f16) @ W[N,K]^T (f16)
//   Block tile 128x128, K-step 32, 256 threads = 8 waves.
//   Wave (wm = w&3, wn = w>>2) computes a 32x64 sub-tile: 2 M-frags x 4 N-frags.
//   A/B tiles staged in LDS (each 128 rows x 32 halfs), double buffered with
//   GLOBAL_LOAD_ASYNC_TO_LDS_B128; each thread issues 4 async b128s per stage.
// Fragment layouts used here (contiguous k, same order for A and B):
//   A: row = lane&15, K = 16*(lane>=16) + {0..15}        -> two 16B LDS loads
//   B: col = lane&15, K = 16*(lane>=16) + {0..15}        -> 32B contiguous
// ---------------------------------------------------------------------------
__global__ __launch_bounds__(256)
void gemm_lds_f16(const _Float16* __restrict__ A,
                  const _Float16* __restrict__ W,
                  float* __restrict__ C,
                  int M, int N, int K) {
    __shared__ __align__(16) _Float16 sA[2][128 * 32];
    __shared__ __align__(16) _Float16 sB[2][128 * 32];

    const int Nt    = N >> 7;
    const int tileM = blockIdx.x / Nt;
    const int tileN = blockIdx.x % Nt;

    const int tid  = threadIdx.x;
    const int w    = tid >> 5;
    const int lane = tid & 31;
    const int wm   = w & 3;    // 4 M-waves
    const int wn   = w >> 2;   // 2 N-waves
    const int half = lane >> 4;
    const int lm   = lane & 15;

    const _Float16* Abase = A + (size_t)(tileM * 128) * K;
    const _Float16* Wbase = W + (size_t)(tileN * 128) * K;

    // per-thread copy slots: 512 chunks (16B) per tile, 2 per thread per tile
    const int c0   = tid;
    const int c1   = tid + 256;
    const int r0   = c0 >> 2, o0 = (c0 & 3) * 8;
    const int r1   = c1 >> 2, o1 = (c1 & 3) * 8;

    v8f acc[2][4];
#pragma unroll
    for (int jm = 0; jm < 2; ++jm)
#pragma unroll
        for (int jn = 0; jn < 4; ++jn) acc[jm][jn] = v8f{};

    const int nsteps = K >> 5;

    // prologue: stage 0
    {
        cp16_g2s(Abase + (size_t)r0 * K + o0, &sA[0][r0 * 32 + o0]);
        cp16_g2s(Abase + (size_t)r1 * K + o1, &sA[0][r1 * 32 + o1]);
        cp16_g2s(Wbase + (size_t)r0 * K + o0, &sB[0][r0 * 32 + o0]);
        cp16_g2s(Wbase + (size_t)r1 * K + o1, &sB[0][r1 * 32 + o1]);
    }

    for (int s = 0; s < nsteps; ++s) {
        const int cur = s & 1;
        if (s + 1 < nsteps) {
            const int nk = (s + 1) * 32;
            cp16_g2s(Abase + (size_t)r0 * K + nk + o0, &sA[cur ^ 1][r0 * 32 + o0]);
            cp16_g2s(Abase + (size_t)r1 * K + nk + o1, &sA[cur ^ 1][r1 * 32 + o1]);
            cp16_g2s(Wbase + (size_t)r0 * K + nk + o0, &sB[cur ^ 1][r0 * 32 + o0]);
            cp16_g2s(Wbase + (size_t)r1 * K + nk + o1, &sB[cur ^ 1][r1 * 32 + o1]);
            async_wait_4();   // previous stage complete, next stage in flight
        } else {
            async_wait_0();
        }
        __syncthreads();

        // A fragments for this wave's two 16-row sub-tiles
        V16 af[2];
#pragma unroll
        for (int jm = 0; jm < 2; ++jm) {
            const _Float16* ap =
                &sA[cur][(wm * 32 + jm * 16 + lm) * 32 + half * 16];
            af[jm].h[0] = *(const v8h*)(ap);
            af[jm].h[1] = *(const v8h*)(ap + 8);
        }
#pragma unroll
        for (int jn = 0; jn < 4; ++jn) {
            const _Float16* bp =
                &sB[cur][(wn * 64 + jn * 16 + lm) * 32 + half * 16];
            V16 b;
            b.h[0] = ((const v8h*)bp)[0];
            b.h[1] = ((const v8h*)bp)[1];
#pragma unroll
            for (int jm = 0; jm < 2; ++jm)
                acc[jm][jn] = wmma_f16(af[jm].v, b.v, acc[jm][jn]);
        }
        __syncthreads();   // everyone done reading buf 'cur' before re-fill
    }

    // epilogue: C/D layout row = i + 8*half, col = lane&15
#pragma unroll
    for (int jm = 0; jm < 2; ++jm) {
#pragma unroll
        for (int jn = 0; jn < 4; ++jn) {
            float* crow = C
                + (size_t)(tileM * 128 + wm * 32 + jm * 16 + 8 * half) * N
                + tileN * 128 + wn * 64 + jn * 16 + lm;
#pragma unroll
            for (int i = 0; i < 8; ++i) crow[(size_t)i * N] = acc[jm][jn][i];
        }
    }
}

// ---------------------------------------------------------------------------
// RoPE: read f32 proj, apply rotation, write f16.
// Layout [S, NH*64]; pairs (d, d+32) within each head, angle = s * inv_freq[d].
// ---------------------------------------------------------------------------
__global__ void rope_f32_to_f16(const float* __restrict__ Qf,
                                _Float16* __restrict__ Qh,
                                const float* __restrict__ inv_freq,
                                int S, int NH) {
    int idx = blockIdx.x * blockDim.x + threadIdx.x;
    int total = S * NH * 32;
    if (idx >= total) return;
    int d  = idx & 31;
    int t  = idx >> 5;
    int hh = t % NH;
    int s  = t / NH;
    int cols = NH * 64;
    size_t b = (size_t)s * cols + hh * 64 + d;
    float ang = (float)s * inv_freq[d];
    float cs = cosf(ang), sn = sinf(ang);
    float x0 = Qf[b], x1 = Qf[b + 32];
    Qh[b]      = (_Float16)(x0 * cs - x1 * sn);
    Qh[b + 32] = (_Float16)(x1 * cs + x0 * sn);
}

// ---------------------------------------------------------------------------
// Transpose V [S, Dv] f32 -> Vt [Dv, S] f16 (so PV WMMA B-loads are contiguous)
// ---------------------------------------------------------------------------
__global__ void transpose_v_f16(const float* __restrict__ Vf,
                                _Float16* __restrict__ Vt, int S, int Dv) {
    int idx = blockIdx.x * blockDim.x + threadIdx.x;
    int total = S * Dv;
    if (idx >= total) return;
    int d = idx % Dv;
    int t = idx / Dv;
    Vt[(size_t)d * S + t] = (_Float16)Vf[(size_t)t * Dv + d];
}

// ---------------------------------------------------------------------------
// Flash attention (causal, GQA). 4 waves/block; each wave owns 16 query rows.
// Per key block of 32: 4 score WMMAs (Kdim=64), online softmax, P through
// per-wave LDS tile (C-layout -> A-layout), 4 PV WMMAs (HD=64).
// ---------------------------------------------------------------------------
__global__ __launch_bounds__(128)
void flash_attn(const _Float16* __restrict__ Qh,
                const _Float16* __restrict__ Kh,
                const _Float16* __restrict__ Vt,
                _Float16* __restrict__ Oh, int S) {
    const int H = 32, KV = 8, HD = 64, G = 4;
    const int Dq = H * HD;   // 2048
    const int Dk = KV * HD;  // 512

    const int qblocks = S >> 6;
    const int h  = blockIdx.x / qblocks;
    const int qb = blockIdx.x % qblocks;
    const int w  = threadIdx.x >> 5;
    const int q0 = qb * 64 + w * 16;
    const int kv = h / G;

    const int lane = threadIdx.x & 31;
    const int half = lane >> 4;
    const int lm   = lane & 15;

    __shared__ __align__(16) _Float16 plds[4][16 * 32];
    _Float16* P = plds[w];

    // Q fragment: A-layout, Kdim = 64 -> two v16h fragments
    V16 aLo, aHi;
    {
        const _Float16* qp = Qh + (size_t)(q0 + lm) * Dq + h * HD + half * 16;
        aLo.h[0] = *(const v8h*)(qp);
        aLo.h[1] = *(const v8h*)(qp + 8);
        aHi.h[0] = *(const v8h*)(qp + 32);
        aHi.h[1] = *(const v8h*)(qp + 40);
    }

    v8f o[4] = {v8f{}, v8f{}, v8f{}, v8f{}};
    float mrow[8], lrow[8];
#pragma unroll
    for (int i = 0; i < 8; ++i) { mrow[i] = -1e30f; lrow[i] = 0.f; }

    for (int t0 = 0; t0 <= q0 + 15; t0 += 32) {
        // ---- scores: S = Q @ K^T, two 16-col tiles --------------------
        v8f s0 = {}, s1 = {};
        {
            const _Float16* kp0 = Kh + (size_t)(t0 + lm) * Dk + kv * HD + half * 16;
            const _Float16* kp1 = Kh + (size_t)(t0 + 16 + lm) * Dk + kv * HD + half * 16;
            V16 b;
            b.h[0] = ((const v8h*)kp0)[0];  b.h[1] = ((const v8h*)kp0)[1];
            s0 = wmma_f16(aLo.v, b.v, s0);
            b.h[0] = ((const v8h*)(kp0 + 32))[0];  b.h[1] = ((const v8h*)(kp0 + 32))[1];
            s0 = wmma_f16(aHi.v, b.v, s0);
            b.h[0] = ((const v8h*)kp1)[0];  b.h[1] = ((const v8h*)kp1)[1];
            s1 = wmma_f16(aLo.v, b.v, s1);
            b.h[0] = ((const v8h*)(kp1 + 32))[0];  b.h[1] = ((const v8h*)(kp1 + 32))[1];
            s1 = wmma_f16(aHi.v, b.v, s1);
        }

        // ---- scale, mask, online softmax ------------------------------
#pragma unroll
        for (int i = 0; i < 8; ++i) {
            float v0 = s0[i] * 0.125f;   // 1/sqrt(64)
            float v1 = s1[i] * 0.125f;
            const int qrow = q0 + i + 8 * half;
            if (t0 + lm      > qrow) v0 = -1e30f;
            if (t0 + 16 + lm > qrow) v1 = -1e30f;

            float rm = fmaxf(v0, v1);
#pragma unroll
            for (int off = 1; off < 16; off <<= 1)
                rm = fmaxf(rm, __shfl_xor(rm, off, 32));

            float mn    = fmaxf(mrow[i], rm);
            float alpha = __expf(mrow[i] - mn);
            mrow[i] = mn;

            float p0 = __expf(v0 - mn), p1 = __expf(v1 - mn);
            float rs = p0 + p1;
#pragma unroll
            for (int off = 1; off < 16; off <<= 1)
                rs += __shfl_xor(rs, off, 32);

            lrow[i] = lrow[i] * alpha + rs;
            o[0][i] *= alpha; o[1][i] *= alpha; o[2][i] *= alpha; o[3][i] *= alpha;

            const int r = i + 8 * half;            // C-layout row
            P[r * 32 + lm]      = (_Float16)p0;
            P[r * 32 + 16 + lm] = (_Float16)p1;
        }

        // ---- reload P in A-layout (in-order DS ops within this wave) ---
        V16 pa;
        {
            const _Float16* pp = P + lm * 32 + half * 16;
            pa.h[0] = *(const v8h*)pp;
            pa.h[1] = *(const v8h*)(pp + 8);
        }

        // ---- O += P @ V (HD = 4 tiles of 16) ---------------------------
#pragma unroll
        for (int nt = 0; nt < 4; ++nt) {
            const _Float16* vp =
                Vt + (size_t)(kv * HD + nt * 16 + lm) * S + t0 + half * 16;
            V16 b;
            b.h[0] = ((const v8h*)vp)[0];
            b.h[1] = ((const v8h*)vp)[1];
            o[nt] = wmma_f16(pa.v, b.v, o[nt]);
        }
    }

    // ---- epilogue: normalize and store f16 attn output -----------------
#pragma unroll
    for (int nt = 0; nt < 4; ++nt) {
        _Float16* op = Oh + (size_t)(q0 + 8 * half) * Dq + h * HD + nt * 16 + lm;
#pragma unroll
        for (int i = 0; i < 8; ++i) {
            float inv = 1.0f / lrow[i];
            op[(size_t)i * Dq] = (_Float16)(o[nt][i] * inv);
        }
    }
}

// ---------------------------------------------------------------------------
// Host launcher
// ---------------------------------------------------------------------------
extern "C" void kernel_launch(void* const* d_in, const int* in_sizes, int n_in,
                              void* d_out, int out_size, void* d_ws, size_t ws_size,
                              hipStream_t stream) {
    const int S = 2048, D = 2048, H = 32, KV = 8, HD = 64;
    const int Dq = H * HD;   // 2048
    const int Dk = KV * HD;  // 512

    const float* x        = (const float*)d_in[0];
    const float* Wq       = (const float*)d_in[1];
    const float* Wk       = (const float*)d_in[2];
    const float* Wv       = (const float*)d_in[3];
    const float* Wo       = (const float*)d_in[4];
    const float* inv_freq = (const float*)d_in[5];

    char* ws = (char*)d_ws;
    size_t off = 0;
    auto take = [&](size_t bytes) -> char* {
        char* p = ws + off;
        off += (bytes + 255) & ~(size_t)255;
        return p;
    };

    _Float16* Xh  = (_Float16*)take((size_t)S * D * 2);
    _Float16* Wqh = (_Float16*)take((size_t)Dq * D * 2);
    _Float16* Wkh = (_Float16*)take((size_t)Dk * D * 2);
    _Float16* Wvh = (_Float16*)take((size_t)Dk * D * 2);
    _Float16* Woh = (_Float16*)take((size_t)D * Dq * 2);
    float*    Qf  = (float*)   take((size_t)S * Dq * 4);
    float*    Kf  = (float*)   take((size_t)S * Dk * 4);
    float*    Vf  = (float*)   take((size_t)S * Dk * 4);
    _Float16* Qhh = (_Float16*)take((size_t)S * Dq * 2);
    _Float16* Khh = (_Float16*)take((size_t)S * Dk * 2);
    _Float16* Vth = (_Float16*)take((size_t)Dk * S * 2);
    _Float16* Ohh = (_Float16*)take((size_t)S * Dq * 2);

    const int T = 256;
    // 1) f32 -> f16 conversions
    cvt_f32_f16<<<(S * D + T - 1) / T, T, 0, stream>>>(x, Xh, S * D);
    cvt_f32_f16<<<(Dq * D + T - 1) / T, T, 0, stream>>>(Wq, Wqh, Dq * D);
    cvt_f32_f16<<<(Dk * D + T - 1) / T, T, 0, stream>>>(Wk, Wkh, Dk * D);
    cvt_f32_f16<<<(Dk * D + T - 1) / T, T, 0, stream>>>(Wv, Wvh, Dk * D);
    cvt_f32_f16<<<(D * Dq + T - 1) / T, T, 0, stream>>>(Wo, Woh, D * Dq);

    // 2) QKV projections (double-buffered LDS WMMA GEMMs), f32 out
    gemm_lds_f16<<<(S / 128) * (Dq / 128), 256, 0, stream>>>(Xh, Wqh, Qf, S, Dq, D);
    gemm_lds_f16<<<(S / 128) * (Dk / 128), 256, 0, stream>>>(Xh, Wkh, Kf, S, Dk, D);
    gemm_lds_f16<<<(S / 128) * (Dk / 128), 256, 0, stream>>>(Xh, Wvh, Vf, S, Dk, D);

    // 3) RoPE -> f16 Q/K ; V transpose -> f16
    rope_f32_to_f16<<<(S * H * 32 + T - 1) / T, T, 0, stream>>>(Qf, Qhh, inv_freq, S, H);
    rope_f32_to_f16<<<(S * KV * 32 + T - 1) / T, T, 0, stream>>>(Kf, Khh, inv_freq, S, KV);
    transpose_v_f16<<<(S * Dk + T - 1) / T, T, 0, stream>>>(Vf, Vth, S, Dk);

    // 4) causal GQA flash attention
    flash_attn<<<H * (S / 64), 128, 0, stream>>>(Qhh, Khh, Vth, Ohh, S);

    // 5) output projection -> f32 d_out
    gemm_lds_f16<<<(S / 128) * (D / 128), 256, 0, stream>>>(Ohh, Woh, (float*)d_out, S, D, Dq);
}
